// SimpleSNN_39359080300690
// MI455X (gfx1250) — matrix operand order repaired and checked by reading, not compiled, not run on hardware
//
#include <hip/hip_runtime.h>

// ---------------- problem constants ----------------
#define T_STEPS 100
#define BATCH   2048
#define IN_F    784
#define HID     128
#define OUTF    10
#define KPAD    800          // 784 padded to 25 * 32 for f16 WMMA K-steps
#define KB1     (KPAD / 32)  // 25 K-steps for layer 1
#define KB2     (HID / 32)   // 4  K-steps for layer 2
#define MT      16           // batch rows per workgroup (one WMMA M tile)
#define NFJ     13           // ceil(16*196 float4 / 256 threads)

typedef __attribute__((ext_vector_type(16))) _Float16 v16h;
typedef __attribute__((ext_vector_type(8)))  _Float16 v8h;
typedef __attribute__((ext_vector_type(8)))  float    v8f;
typedef __attribute__((ext_vector_type(4)))  float    v4f;

// ---------------- LDS layout (bytes), all 16B aligned ----------------
#define SA0_OFF   0                          // spikes buf0: 16 x 800 f16 = 25600
#define SA1_OFF   (SA0_OFF   + MT*KPAD*2)    // spikes buf1:               25600
#define SP1A_OFF  (SA1_OFF   + MT*KPAD*2)    // sp1 buf0: 16 x 128 f16   =  4096
#define SP1B_OFF  (SP1A_OFF  + MT*HID*2)     // sp1 buf1:                   4096
#define SW2_OFF   (SP1B_OFF  + MT*HID*2)     // W2^T: 16 x 128 f16       =  4096
#define SPROB_OFF (SW2_OFF   + 16*HID*2)     // prob: 16 x 784 f32       = 50176
#define SW1_OFF   (SPROB_OFF + MT*IN_F*4)    // W1^T: 128 x 800 f16      = 204800
#define SMEM_BYTES (SW1_OFF + HID*KPAD*2)    // = 318464  (<= 320KB/WGP)

// ---------------- WMMA helpers ----------------
__device__ __forceinline__ v8f wmma_f16(v16h a, v16h b, v8f c) {
  // v_wmma_f32_16x16x32_f16: (neg_a, A, neg_b, B, c_mod, C, reuse_a, reuse_b)
  return __builtin_amdgcn_wmma_f32_16x16x32_f16(false, a, false, b,
                                                (short)0, c, false, false);
}

// Fragment loads from a per-lane base pointer; k is a compile-time-constant
// half-element offset so both 16B ds_load_b128 use folded offset immediates.
// A (16-bit 16x32): lane already positioned at row m=lane&15, +8 halfs for
// high lanes; element halves are [k .. k+7] and [k+16 .. k+23].
__device__ __forceinline__ v16h frag_a_at(const _Float16* p, int k) {
  v8h lo8 = *(const v8h*)(p + k);
  v8h hi8 = *(const v8h*)(p + k + 16);
  return __builtin_shufflevector(lo8, hi8, 0,1,2,3,4,5,6,7,8,9,10,11,12,13,14,15);
}
// B (16-bit 32x16, tile stored transposed [n][k]): lane positioned at column
// n=lane&15, +8 halfs for high lanes; element halves are [k .. k+7] and [k+16 .. k+23].
__device__ __forceinline__ v16h frag_b_at(const _Float16* p, int k) {
  v8h lo8 = *(const v8h*)(p + k);
  v8h hi8 = *(const v8h*)(p + k + 16);
  return __builtin_shufflevector(lo8, hi8, 0,1,2,3,4,5,6,7,8,9,10,11,12,13,14,15);
}

// Non-temporal stream of one enc_noise tile into registers (one-pass data:
// NT hint keeps L2 free for W1/x which every workgroup re-reads).
__device__ __forceinline__ void prefetch_enc(v4f* rbuf, const float* enc,
                                             int t, int b0, int tid) {
  const size_t base = (size_t)t * ((size_t)BATCH * IN_F) + (size_t)b0 * IN_F;
  #pragma unroll
  for (int j = 0; j < NFJ; ++j) {
    int idx = tid + j * 256;
    if (idx < MT * (IN_F / 4)) {
      int m = idx / (IN_F / 4), c4 = idx - m * (IN_F / 4);
      rbuf[j] = __builtin_nontemporal_load(
          (const v4f*)(enc + base + (size_t)m * IN_F + c4 * 4));
    }
  }
}

// Bernoulli spikes: select f16 1.0 bit pattern (0x3C00) directly -- no cvt.
__device__ __forceinline__ void spikes_to_lds(const v4f* rbuf, const float* sProb,
                                              _Float16* sAbuf, int tid) {
  #pragma unroll
  for (int j = 0; j < NFJ; ++j) {
    int idx = tid + j * 256;
    if (idx < MT * (IN_F / 4)) {
      int m = idx / (IN_F / 4), c4 = idx - m * (IN_F / 4);
      v4f e = rbuf[j];
      v4f p = *(const v4f*)(sProb + m * IN_F + c4 * 4);
      unsigned lo = (e.x < p.x ? 0x00003C00u : 0u) | (e.y < p.y ? 0x3C000000u : 0u);
      unsigned hi = (e.z < p.z ? 0x00003C00u : 0u) | (e.w < p.w ? 0x3C000000u : 0u);
      uint2 s; s.x = lo; s.y = hi;
      *(uint2*)(sAbuf + m * KPAD + c4 * 4) = s;   // ds_store_b64
    }
  }
}

extern "C" __global__ void __launch_bounds__(256, 1)
snn_lif_wmma(const float* __restrict__ x,   const float* __restrict__ enc,
             const float* __restrict__ W1,  const float* __restrict__ b1,
             const float* __restrict__ W2,  const float* __restrict__ b2,
             float* __restrict__ out)
{
  extern __shared__ char smem[];
  _Float16* sA0   = (_Float16*)(smem + SA0_OFF);
  _Float16* sA1   = (_Float16*)(smem + SA1_OFF);
  _Float16* sP1a  = (_Float16*)(smem + SP1A_OFF);
  _Float16* sP1b  = (_Float16*)(smem + SP1B_OFF);
  _Float16* sW2   = (_Float16*)(smem + SW2_OFF);
  float*    sProb = (float*)   (smem + SPROB_OFF);
  _Float16* sW1   = (_Float16*)(smem + SW1_OFF);

  const int tid  = threadIdx.x;
  const int lane = tid & 31;
  const int wave = tid >> 5;
  const int b0   = blockIdx.x * MT;

  // Kick off the t=0 enc stream first so it overlaps all LDS setup.
  v4f rbuf[NFJ];
  prefetch_enc(rbuf, enc, 0, b0, tid);

  // ---------------- one-time setup ----------------
  // W1 [784][128] f32 -> LDS f16 transposed [n][k], K padded to 800.
  for (int idx = tid; idx < IN_F * HID; idx += 256) {
    int k = idx >> 7, n = idx & 127;
    sW1[n * KPAD + k] = (_Float16)W1[idx];
  }
  for (int idx = tid; idx < HID * (KPAD - IN_F); idx += 256) {
    int n = idx >> 4;
    sW1[n * KPAD + IN_F + (idx & 15)] = (_Float16)0.0f;
  }
  // W2 [128][10] -> LDS f16 transposed [n][k], N padded 10->16 with zeros.
  for (int idx = tid; idx < 16 * HID; idx += 256) {
    int n = idx >> 7, k = idx & 127;
    sW2[n * HID + k] = (n < OUTF) ? (_Float16)W2[k * OUTF + n] : (_Float16)0.0f;
  }
  // Poisson rate: prob = clip(x * 0.1, 0, 1), staged once in LDS.
  for (int idx = tid; idx < MT * IN_F; idx += 256) {
    int m = idx / IN_F, c = idx - m * IN_F;
    float p = x[(size_t)(b0 + m) * IN_F + c] * 0.1f;
    p = p < 0.f ? 0.f : (p > 1.f ? 1.f : p);
    sProb[m * IN_F + c] = p;
  }
  // Zero the K padding of both spike buffers once (never rewritten).
  for (int idx = tid; idx < MT * (KPAD - IN_F); idx += 256) {
    int off = (idx >> 4) * KPAD + IN_F + (idx & 15);
    sA0[off] = (_Float16)0.0f;
    sA1[off] = (_Float16)0.0f;
  }

  // Per-wave constants: wave owns hidden columns [n0, n0+16).
  const int   n0    = wave << 4;
  const float bias1 = b1[n0 + (lane & 15)];
  const float bias2 = ((lane & 15) < OUTF) ? b2[lane & 15] : 0.f;

  // Per-lane LDS base pointers (fold row/column/k-group terms once).
  const int laneM = (lane & 15);
  const int aHi   = (lane >> 4) << 3;     // A: +8 halfs on high half-wave
  const int bHi   = (lane >> 4) << 3;     // B: +8 halfs on high half-wave
  const _Float16* paA0 = sA0  + laneM * KPAD + aHi;
  const _Float16* paA1 = sA1  + laneM * KPAD + aHi;
  const _Float16* pbW1 = sW1  + n0 * KPAD + laneM * KPAD + bHi;
  const _Float16* paP0 = sP1a + laneM * HID + aHi;
  const _Float16* paP1 = sP1b + laneM * HID + aHi;
  const _Float16* pbW2 = sW2  + laneM * HID + bHi;

  v8f mem1, mem2, counts;
  #pragma unroll
  for (int r = 0; r < 8; ++r) { mem1[r] = 0.f; mem2[r] = 0.f; counts[r] = 0.f; }

  __syncthreads();                       // setup (sProb/sW1/sW2/padding) visible
  spikes_to_lds(rbuf, sProb, sA0, tid);  // spikes for t = 0 -> buffer 0
  __syncthreads();

  // -------- time loop: ONE barrier per step, double-buffered sA / sp1 -------
  for (int t = 0; t < T_STEPS; ++t) {
    const _Float16* pa   = (t & 1) ? paA1 : paA0;
    _Float16*       sAnxt= (t & 1) ? sA0  : sA1;
    const _Float16* pa2  = (t & 1) ? paP1 : paP0;
    _Float16*       sPcur= (t & 1) ? sP1b : sP1a;
    const bool more = (t + 1 < T_STEPS);

    // (a) stream next enc tile; latency hidden behind the WMMA phase below
    if (more) prefetch_enc(rbuf, enc, t + 1, b0, tid);

    // (b) LIF layer 1:  mem1 = 0.9*mem1 + S @ W1 + b1
    //     3-slot rotated software pipeline: loads for kb+2 issue before the
    //     WMMA of kb, keeping 8 ds_load_b128 in flight per wait.
    v8f acc;
    #pragma unroll
    for (int r = 0; r < 8; ++r) acc[r] = mem1[r] * 0.9f;

    v16h af[3], bf[3];
    af[0] = frag_a_at(pa, 0);     bf[0] = frag_b_at(pbW1, 0);
    af[1] = frag_a_at(pa, 32);    bf[1] = frag_b_at(pbW1, 32);
    #pragma unroll
    for (int kb = 0; kb < KB1; ++kb) {
      if (kb + 2 < KB1) {
        af[(kb + 2) % 3] = frag_a_at(pa,   (kb + 2) * 32);
        bf[(kb + 2) % 3] = frag_b_at(pbW1, (kb + 2) * 32);
      }
      acc = wmma_f16(af[kb % 3], bf[kb % 3], acc);
    }
    {
      const int mrow = (lane >> 4) << 3;   // C/D layout: M = r + 8*(lane>=16)
      const int ncol = lane & 15;
      #pragma unroll
      for (int r = 0; r < 8; ++r) {
        float v  = acc[r] + bias1;
        float sp = (v >= 1.0f) ? 1.0f : 0.0f;
        mem1[r]  = v * (1.0f - sp);                       // reset-to-zero
        sPcur[(mrow + r) * HID + n0 + ncol] = (_Float16)sp;
      }
    }

    // (c) spikes for t+1 into the other buffer (overlaps other waves' WMMA)
    if (more) spikes_to_lds(rbuf, sProb, sAnxt, tid);

    // (d) single barrier: publishes sp1(t) and spikes(t+1)
    __syncthreads();

    // (e) LIF layer 2 on wave 0 (uniform branch; EXEC all-1s inside the wave).
    //     Runs concurrently with waves 1..7 starting iteration t+1.
    if (wave == 0) {
      v8f acc2;
      #pragma unroll
      for (int r = 0; r < 8; ++r) acc2[r] = mem2[r] * 0.9f;
      v16h a2[KB2], b2f[KB2];
      #pragma unroll
      for (int kb = 0; kb < KB2; ++kb) {   // issue all 16 loads first
        a2[kb]  = frag_a_at(pa2, kb * 32);
        b2f[kb] = frag_b_at(pbW2, kb * 32);
      }
      #pragma unroll
      for (int kb = 0; kb < KB2; ++kb)
        acc2 = wmma_f16(a2[kb], b2f[kb], acc2);
      #pragma unroll
      for (int r = 0; r < 8; ++r) {
        float v  = acc2[r] + bias2;
        float sp = (v >= 1.0f) ? 1.0f : 0.0f;
        mem2[r]  = v * (1.0f - sp);
        counts[r] += sp;
      }
    }
  }

  // ---------------- rate decode: write spike counts ----------------
  if (wave == 0) {
    const int ncol = lane & 15;
    const int mrow = (lane >> 4) << 3;
    if (ncol < OUTF) {
      #pragma unroll
      for (int r = 0; r < 8; ++r)
        out[(size_t)(b0 + mrow + r) * OUTF + ncol] = counts[r];
    }
  }
}

extern "C" void kernel_launch(void* const* d_in, const int* in_sizes, int n_in,
                              void* d_out, int out_size, void* d_ws, size_t ws_size,
                              hipStream_t stream) {
  const float* x   = (const float*)d_in[0];
  const float* enc = (const float*)d_in[1];
  const float* W1  = (const float*)d_in[2];
  const float* b1  = (const float*)d_in[3];
  const float* W2  = (const float*)d_in[4];
  const float* b2  = (const float*)d_in[5];
  float* out = (float*)d_out;

  dim3 grid(BATCH / MT);   // 128 workgroups, one 16-row batch tile each
  dim3 block(256);         // 8 wave32s
  snn_lif_wmma<<<grid, block, SMEM_BYTES, stream>>>(x, enc, W1, b1, W2, b2, out);
}
